// MultiHeadAttention_15015205667122
// MI455X (gfx1250) — compile-verified, matrix operand order checked by reading, not run on hardware
//
#include <hip/hip_runtime.h>
#include <hip/hip_bf16.h>

// MHA for B=4, S=2048, M=1024, H=16, D=64 on gfx1250 (wave32, WMMA f16->f32).
//
// Pipeline:
//   1) cvt_t_kernel: Wq/Wk/Wv [H][M][D] f32 -> [H][D][M] f16 ; Wo -> transposed f16
//   2) proj_kernel : Q,K -> [B][H][S][D] f16 ; V -> [B][H][D][S] f16 (transposed)
//   3) flash_kernel: online-softmax attention with double-buffered
//      global_load_async_to_lds_b128 staging of K/V tiles; O -> [B][S][H*D] f16
//   4) outproj_kernel: out = O @ Wo -> [B][S][M] f32

typedef __attribute__((ext_vector_type(16))) _Float16 v16h;
typedef __attribute__((ext_vector_type(8)))  _Float16 v8h;
typedef __attribute__((ext_vector_type(8)))  float    v8f;

#define DEV __device__ __forceinline__

static constexpr int Bc = 4, Sc = 2048, Mc = 1024, Hc = 16, Dc = 64;

union Frag16 { v16h v; v8h h[2]; };

// A-style fragment load for v_wmma_f32_16x16x32_f16 from a row-major f16 matrix.
// 16-bit A layout: lane L (L<16) = row L holds K = {0..7, 16..23};
// lane L+16 = row L holds K = {8..15, 24..31}.
// Per lane: 8 contiguous halves at (c0 + kgrp*8) and 8 more at (+16).
// Applied to a transposed-stored matrix this also yields the B fragment.
DEV v16h load_frag_f16(const _Float16* __restrict__ base, int row, int ld, int c0, int kgrp) {
  const _Float16* p = base + (size_t)row * ld + c0 + kgrp * 8;
  Frag16 f;
  f.h[0] = *(const v8h*)(p);        // j = 0..7
  f.h[1] = *(const v8h*)(p + 16);   // j = 8..15
  return f.v;
}

DEV v16h load_frag_f32cvt(const float* __restrict__ base, int row, int ld, int c0, int kgrp) {
  const float* p = base + (size_t)row * ld + c0 + kgrp * 8;
  v16h r;
#pragma unroll
  for (int j = 0; j < 8; ++j) {
    r[j]     = (_Float16)p[j];
    r[j + 8] = (_Float16)p[j + 16];
  }
  return r;
}

DEV v8f wmma16(v16h a, v16h b, v8f c) {
  return __builtin_amdgcn_wmma_f32_16x16x32_f16(false, a, false, b, (short)0, c, false, false);
}

// Async 16-byte global -> LDS copy (per lane).  Flat LDS aperture uses
// addr[31:0] as the LDS byte address, so truncating the generic pointer to
// 32 bits yields the VDST LDS address for the async-to-LDS encoding.
DEV void async_ld16(const _Float16* g, _Float16* l) {
  unsigned lds_addr = (unsigned)(uintptr_t)l;
  asm volatile("global_load_async_to_lds_b128 %0, %1, off"
               :: "v"(lds_addr), "v"(g)
               : "memory");
}
DEV void wait_async_le2() { asm volatile("s_wait_asynccnt 0x2" ::: "memory"); }
DEV void wait_async_0()   { asm volatile("s_wait_asynccnt 0x0" ::: "memory"); }

// ---------------------------------------------------------------------------
// Convert + transpose: in [nmat][rows][cols] f32 -> out [nmat][cols][rows] f16
// ---------------------------------------------------------------------------
__global__ __launch_bounds__(256) void cvt_t_kernel(const float* __restrict__ in,
                                                    _Float16* __restrict__ out,
                                                    int rows, int cols) {
  size_t mbase = (size_t)blockIdx.z * rows * cols;
  int idx = blockIdx.x * 256 + threadIdx.x;
  if (idx < rows * cols) {
    int r = idx / cols, c = idx - r * cols;
    out[mbase + (size_t)c * rows + r] = (_Float16)in[mbase + idx];
  }
}

// ---------------------------------------------------------------------------
// Projection GEMM: per (b,h) each wave computes a 16x64 tile of X[b] @ W[h].
// X: [B*S][M] f32.  Wt: [H][D][M] f16.  transposeOut: 0 -> [b][h][s][d],
// 1 -> [b][h][d][s].
// ---------------------------------------------------------------------------
__global__ __launch_bounds__(256) void proj_kernel(const float* __restrict__ X,
                                                   const _Float16* __restrict__ Wt,
                                                   _Float16* __restrict__ Out,
                                                   int transposeOut) {
  const int wave = threadIdx.x >> 5, lane = threadIdx.x & 31;
  const int kgrp = lane >> 4, lr = lane & 15;
  const int bh = blockIdx.y;
  const int b = bh >> 4, h = bh & 15;
  const int s0 = blockIdx.x * 128 + wave * 16;

  const float*    xb = X  + (size_t)b * Sc * Mc;
  const _Float16* wt = Wt + (size_t)h * Dc * Mc;   // [D][M]

  v8f acc[4] = {v8f{}, v8f{}, v8f{}, v8f{}};
  for (int k0 = 0; k0 < Mc; k0 += 32) {
    v16h a = load_frag_f32cvt(xb + (size_t)s0 * Mc, lr, Mc, k0, kgrp);
#pragma unroll
    for (int t = 0; t < 4; ++t) {
      v16h bf = load_frag_f16(wt + (size_t)(t * 16) * Mc, lr, Mc, k0, kgrp);
      acc[t] = wmma16(a, bf, acc[t]);
    }
  }
#pragma unroll
  for (int t = 0; t < 4; ++t) {
#pragma unroll
    for (int i = 0; i < 8; ++i) {
      int s = s0 + i + 8 * kgrp;
      int d = t * 16 + lr;
      _Float16 val = (_Float16)acc[t][i];
      if (!transposeOut)
        Out[(((size_t)bh) * Sc + s) * Dc + d] = val;            // [b][h][s][d]
      else
        Out[(((size_t)bh) * Dc + d) * Sc + s] = val;            // [b][h][d][s]
    }
  }
}

// ---------------------------------------------------------------------------
// Flash attention.  8 waves/block share one (b,h); each wave owns 16 queries.
// K/V 32-key tiles are staged into LDS once per block with async-to-LDS
// copies (double buffered), then all waves read fragments via ds_load_b128.
// Qh,Kh: [B][H][S][D] f16 ; Vt: [B][H][D][S] f16 ; Oh: [B][S][H*D] f16
// ---------------------------------------------------------------------------
__global__ __launch_bounds__(256) void flash_kernel(const _Float16* __restrict__ Qh,
                                                    const _Float16* __restrict__ Kh,
                                                    const _Float16* __restrict__ Vt,
                                                    _Float16* __restrict__ Oh) {
  // padded row strides (72/40 halves: keep 16B alignment, spread LDS banks)
  __shared__ __align__(16) _Float16 Klds[2][32][72];   // keys x d
  __shared__ __align__(16) _Float16 Vlds[2][64][40];   // d x keys
  __shared__ __align__(16) _Float16 pshm[8][16][40];   // per-wave 16q x 32k probs

  const int tid = threadIdx.x;
  const int wave = tid >> 5, lane = tid & 31;
  const int kgrp = lane >> 4, lr = lane & 15;
  const int bh = blockIdx.y;
  const int b = bh >> 4, h = bh & 15;
  const int q0 = blockIdx.x * 128 + wave * 16;

  const _Float16* Qp = Qh + (size_t)bh * Sc * Dc;
  const _Float16* Kp = Kh + (size_t)bh * Sc * Dc;
  const _Float16* Vp = Vt + (size_t)bh * Dc * Sc;

  // per-thread staging coordinates (256 threads x 16B == one 4KB tile each)
  const int krow = tid >> 3, kcol = (tid & 7) * 8;   // K tile: 32 x 64 halves
  const int vrow = tid >> 2, vcol = (tid & 3) * 8;   // V tile: 64 x 32 halves

  // Q fragments: d-chunks [0,32) and [32,64)
  v16h qa0 = load_frag_f16(Qp + (size_t)q0 * Dc, lr, Dc, 0,  kgrp);
  v16h qa1 = load_frag_f16(Qp + (size_t)q0 * Dc, lr, Dc, 32, kgrp);

  v8f o[4] = {v8f{}, v8f{}, v8f{}, v8f{}};
  float mrow[8], lrow[8];
#pragma unroll
  for (int i = 0; i < 8; ++i) { mrow[i] = -1e30f; lrow[i] = 0.0f; }

  const float cse = 0.125f * 1.44269504089f;   // (1/sqrt(64)) * log2(e)
  const int NT = Sc / 32;

  // stage tile 0 into buffer 0
  async_ld16(Kp + (size_t)krow * Dc + kcol,      &Klds[0][krow][kcol]);
  async_ld16(Vp + (size_t)vrow * Sc + vcol,      &Vlds[0][vrow][vcol]);

  for (int it = 0; it < NT; ++it) {
    const int buf = it & 1;
    const int t0 = it * 32;
    if (it + 1 < NT) {
      const int tn = t0 + 32;
      async_ld16(Kp + (size_t)(tn + krow) * Dc + kcol, &Klds[buf ^ 1][krow][kcol]);
      async_ld16(Vp + (size_t)vrow * Sc + tn + vcol,   &Vlds[buf ^ 1][vrow][vcol]);
      wait_async_le2();   // current tile landed; next tile still in flight
    } else {
      wait_async_0();
    }
    __syncthreads();

    // scores: S[16q x 32k] = Q @ K^T ; B-operand of K^T == A-style load of K rows
    v8f s0 = v8f{}, s1 = v8f{};
#pragma unroll
    for (int dc = 0; dc < 2; ++dc) {
      v16h kb0 = load_frag_f16(&Klds[buf][0][0],  lr, 72, dc * 32, kgrp);
      v16h kb1 = load_frag_f16(&Klds[buf][16][0], lr, 72, dc * 32, kgrp);
      v16h qa  = dc ? qa1 : qa0;
      s0 = wmma16(qa, kb0, s0);
      s1 = wmma16(qa, kb1, s1);
    }

    // online softmax; row M = i + 8*kgrp lives entirely in this 16-lane half
    float alpha[8];
#pragma unroll
    for (int i = 0; i < 8; ++i) {
      float mx = fmaxf(s0[i], s1[i]);
      for (int off = 8; off >= 1; off >>= 1) mx = fmaxf(mx, __shfl_xor(mx, off, 32));
      float m_new = fmaxf(mrow[i], mx * cse);
      alpha[i] = exp2f(mrow[i] - m_new);
      mrow[i] = m_new;
      float p0 = exp2f(s0[i] * cse - m_new);
      float p1 = exp2f(s1[i] * cse - m_new);
      float ps = p0 + p1;
      for (int off = 8; off >= 1; off >>= 1) ps += __shfl_xor(ps, off, 32);
      lrow[i] = lrow[i] * alpha[i] + ps;
      pshm[wave][i + 8 * kgrp][lr]      = (_Float16)p0;   // key = lr
      pshm[wave][i + 8 * kgrp][lr + 16] = (_Float16)p1;   // key = 16 + lr
    }
    __syncthreads();
    v16h pa = load_frag_f16(&pshm[wave][0][0], lr, 40, 0, kgrp);

    // O = O*alpha + P @ V ; B-operand of V == A-style load of Vt rows (d)
#pragma unroll
    for (int t = 0; t < 4; ++t) {
#pragma unroll
      for (int i = 0; i < 8; ++i) o[t][i] *= alpha[i];
      v16h vb = load_frag_f16(&Vlds[buf][t * 16][0], lr, 40, 0, kgrp);
      o[t] = wmma16(pa, vb, o[t]);
    }
    __syncthreads();   // all waves done with this buffer before it is re-staged
  }

  // epilogue: O /= l ; store Oh[b][s][h*64 + d]
#pragma unroll
  for (int t = 0; t < 4; ++t) {
#pragma unroll
    for (int i = 0; i < 8; ++i) {
      int s = q0 + i + 8 * kgrp;
      int d = t * 16 + lr;
      float val = o[t][i] / lrow[i];
      Oh[((size_t)b * Sc + s) * (Hc * Dc) + h * Dc + d] = (_Float16)val;
    }
  }
}

// ---------------------------------------------------------------------------
// Output projection: out[BS x 1024] = Oh[BS x 1024] @ Wo ; WoT: [n][k] f16
// ---------------------------------------------------------------------------
__global__ __launch_bounds__(256) void outproj_kernel(const _Float16* __restrict__ Oh,
                                                      const _Float16* __restrict__ WoT,
                                                      float* __restrict__ Out) {
  const int wave = threadIdx.x >> 5, lane = threadIdx.x & 31;
  const int kgrp = lane >> 4, lr = lane & 15;
  const int r0 = blockIdx.x * 128 + wave * 16;   // row in [0, B*S)
  const int n0 = blockIdx.y * 64;                // col in [0, M)
  const int Kd = Hc * Dc;                        // 1024

  v8f acc[4] = {v8f{}, v8f{}, v8f{}, v8f{}};
  for (int k0 = 0; k0 < Kd; k0 += 32) {
    v16h a = load_frag_f16(Oh + (size_t)r0 * Kd, lr, Kd, k0, kgrp);
#pragma unroll
    for (int t = 0; t < 4; ++t) {
      v16h bf = load_frag_f16(WoT + (size_t)(n0 + t * 16) * Kd, lr, Kd, k0, kgrp);
      acc[t] = wmma16(a, bf, acc[t]);
    }
  }
#pragma unroll
  for (int t = 0; t < 4; ++t)
#pragma unroll
    for (int i = 0; i < 8; ++i)
      Out[(size_t)(r0 + i + 8 * kgrp) * Mc + n0 + t * 16 + lr] = acc[t][i];
}

// ---------------------------------------------------------------------------
extern "C" void kernel_launch(void* const* d_in, const int* in_sizes, int n_in,
                              void* d_out, int out_size, void* d_ws, size_t ws_size,
                              hipStream_t stream) {
  (void)in_sizes; (void)n_in; (void)out_size; (void)ws_size;
  const float* q  = (const float*)d_in[0];
  const float* k  = (const float*)d_in[1];
  const float* v  = (const float*)d_in[2];
  const float* Wq = (const float*)d_in[3];
  const float* Wk = (const float*)d_in[4];
  const float* Wv = (const float*)d_in[5];
  const float* Wo = (const float*)d_in[6];
  float* out = (float*)d_out;

  char* ws = (char*)d_ws;
  _Float16* Qh  = (_Float16*)(ws + (0ull  << 20));  // 16 MiB  [B][H][S][D]
  _Float16* Kh  = (_Float16*)(ws + (16ull << 20));  // 16 MiB  [B][H][S][D]
  _Float16* Vt  = (_Float16*)(ws + (32ull << 20));  // 16 MiB  [B][H][D][S]
  _Float16* Oh  = (_Float16*)(ws + (48ull << 20));  // 16 MiB  [B][S][H*D]
  _Float16* WqT = (_Float16*)(ws + (64ull << 20));  //  2 MiB  [H][D][M]
  _Float16* WkT = (_Float16*)(ws + (66ull << 20));  //  2 MiB
  _Float16* WvT = (_Float16*)(ws + (68ull << 20));  //  2 MiB
  _Float16* WoT = (_Float16*)(ws + (70ull << 20));  //  2 MiB  [M][H*D]

  // 1) weight convert + transpose
  cvt_t_kernel<<<dim3(256, 1, 16),  256, 0, stream>>>(Wq, WqT, Mc, Dc);
  cvt_t_kernel<<<dim3(256, 1, 16),  256, 0, stream>>>(Wk, WkT, Mc, Dc);
  cvt_t_kernel<<<dim3(256, 1, 16),  256, 0, stream>>>(Wv, WvT, Mc, Dc);
  cvt_t_kernel<<<dim3(4096, 1, 1),  256, 0, stream>>>(Wo, WoT, Hc * Dc, Mc);

  // 2) QKV projections (V transposed for the P@V operand)
  dim3 pgrid(Sc / 128, Bc * Hc);
  proj_kernel<<<pgrid, 256, 0, stream>>>(q, WqT, Qh, 0);
  proj_kernel<<<pgrid, 256, 0, stream>>>(k, WkT, Kh, 0);
  proj_kernel<<<pgrid, 256, 0, stream>>>(v, WvT, Vt, 1);

  // 3) flash attention
  flash_kernel<<<pgrid, 256, 0, stream>>>(Qh, Kh, Vt, Oh);

  // 4) output projection
  outproj_kernel<<<dim3((Bc * Sc) / 128, Mc / 64), 256, 0, stream>>>(Oh, WoT, out);
}
